// GAT_LCG_15839839387881
// MI455X (gfx1250) — compile-verified, matrix operand order checked by reading, not run on hardware
//
#include <hip/hip_runtime.h>
#include <hip/hip_bf16.h>

#define GDIM 128
#define NEG_SLOPE 0.2f

typedef _Float16 v16h __attribute__((ext_vector_type(16)));
typedef float    v8f  __attribute__((ext_vector_type(8)));
typedef _Float16 h4   __attribute__((ext_vector_type(4)));

// ---------------------------------------------------------------------------
// order-preserving float <-> u32 map for atomic segment-max
// ---------------------------------------------------------------------------
__device__ __forceinline__ unsigned fmap(float f) {
    unsigned u = __float_as_uint(f);
    return (u & 0x80000000u) ? ~u : (u | 0x80000000u);
}
__device__ __forceinline__ float funmap(unsigned u) {
    unsigned b = (u & 0x80000000u) ? (u ^ 0x80000000u) : ~u;
    return __uint_as_float(b);
}

// ---------------------------------------------------------------------------
// weight repack: f32 row-major [K,128]  ->  f16 WMMA B-fragment order
//   Wp[chunk][tile][lane][16 halves], chunk = k/32, tile = n/16.
//   Lane l of tile t holds column t*16+(l&15); half h=2v+j maps to
//   k = chunk*32 + 8*(l>>4) + (v<4?0:16) + 2*(v&3) + j   (ISA §7.12.2).
// One 32-byte contiguous load per lane per WMMA in the hot loop.
// ---------------------------------------------------------------------------
__global__ void pack_w_kernel(const float* __restrict__ W,
                              _Float16* __restrict__ out, int K) {
    int o = blockIdx.x * blockDim.x + threadIdx.x;
    if (o >= K * GDIM) return;
    int h     = o & 15;
    int lane  = (o >> 4) & 31;
    int tile  = (o >> 9) & 7;
    int chunk = o >> 12;
    int v = h >> 1, j = h & 1;
    int k = chunk * 32 + ((lane >> 4) << 3) + ((v < 4) ? 0 : 16) + 2 * (v & 3) + j;
    int col = tile * 16 + (lane & 15);
    out[o] = (_Float16)W[k * GDIM + col];
}

// ---------------------------------------------------------------------------
// zero a region of 32-bit words
// ---------------------------------------------------------------------------
__global__ void zero32_kernel(unsigned* __restrict__ p, long long n) {
    long long i = (long long)blockIdx.x * blockDim.x + threadIdx.x;
    if (i < n) p[i] = 0u;
}

// ---------------------------------------------------------------------------
// WMMA GEMM:  Y[N,128] = act( concat(X0,X1,X2)[N, 128*NSRC] @ W + bias )
//   Wp is fragment-packed f16 (see pack_w_kernel). xor0: row^1 on source 0
//   (literal <-> negated-literal permutation for the l2l MLP).
// Block = 256 threads = 8 wave32s; block computes a 16-row x 128-col strip,
// wave w owns the 16x16 tile at columns [16w,16w+16).
// A strip staged f32->f16 into LDS in fragment order: per chunk, per lane,
// 16 contiguous halves -> one 32B ds read per WMMA.
// ---------------------------------------------------------------------------
template <int NSRC>
__global__ __launch_bounds__(256)
void gemm_wmma_kernel(const float* __restrict__ X0, const float* __restrict__ X1,
                      const float* __restrict__ X2, int xor0,
                      const _Float16* __restrict__ Wp, const float* __restrict__ bias,
                      float* __restrict__ Y, int N, int do_relu)
{
    constexpr int K  = NSRC * GDIM;   // 128 / 256 / 384
    constexpr int CH = K / 32;        // k-chunks (WMMAs per tile)
    __shared__ __align__(32) _Float16 As[K * 16];   // CH*32 lanes*16 halves

    const int m0  = blockIdx.x * 16;
    const int tid = threadIdx.x;

    // ---- stage A strip (16 rows x K) into LDS fragment order ----
    const int nf4 = (16 * K) >> 2;
    for (int s = tid; s < nf4; s += 256) {
        int idx = s << 2;                 // element index (row-major strip)
        int r   = idx / K;                // 0..15
        int k   = idx - r * K;            // multiple of 4
        int row = m0 + r;
        int src = k >> 7;
        int kk  = k & (GDIM - 1);
        const float* Xs = (src == 0) ? X0 : ((src == 1) ? X1 : X2);
        int rowe = (src == 0 && xor0) ? (row ^ 1) : row;
        float4 f = make_float4(0.f, 0.f, 0.f, 0.f);
        if (row < N) {
            f = *(const float4*)(Xs + (size_t)rowe * GDIM + kk);
            if (row + 16 < N)   // gfx1250 global_prefetch_b8 for next strip
                __builtin_prefetch(Xs + (size_t)(rowe + 16) * GDIM + kk, 0, 1);
        }
        // fragment position: octet mapping (ISA §7.12.2 16-bit A layout)
        int chunk = k >> 5;
        int kc    = k & 31;
        int dlane = r + (((kc >> 3) & 1) << 4);
        int dhb   = (((kc >> 4) & 1) << 3) + (kc & 7);
        h4 t; t[0] = (_Float16)f.x; t[1] = (_Float16)f.y;
              t[2] = (_Float16)f.z; t[3] = (_Float16)f.w;
        *(h4*)(As + (((chunk * 32 + dlane) << 4) + dhb)) = t;
    }
    __syncthreads();

    const int lane = tid & 31;
    const int wv   = tid >> 5;            // column tile
    const int n0   = wv * 16;
    const int colr = lane & 15;

    v8f acc = {0.f, 0.f, 0.f, 0.f, 0.f, 0.f, 0.f, 0.f};
#pragma unroll
    for (int c = 0; c < CH; ++c) {
        v16h a = *(const v16h*)(As + ((c * 32 + lane) << 4));
        v16h b = *(const v16h*)(Wp + ((size_t)(((c * 8 + wv) * 32) + lane) << 4));
        acc = __builtin_amdgcn_wmma_f32_16x16x32_f16(
            /*neg_a=*/false, a, /*neg_b=*/false, b,
            /*c_mod=*/(short)0, acc, /*reuse_a=*/false, /*reuse_b=*/false);
    }

    const float bcol = bias[n0 + colr];
    float* yp = Y + (size_t)(m0 + ((lane >> 4) << 3)) * GDIM + n0 + colr;
    if (m0 + 16 <= N) {                   // uniform fast path: no exec masking
#pragma unroll
        for (int v = 0; v < 8; ++v) {
            float y = acc[v] + bcol;
            if (do_relu) y = fmaxf(y, 0.f);
            yp[(size_t)v * GDIM] = y;
        }
    } else {
#pragma unroll
        for (int v = 0; v < 8; ++v) {
            int row = m0 + v + ((lane >> 4) << 3);
            float y = acc[v] + bcol;
            if (do_relu) y = fmaxf(y, 0.f);
            if (row < N) Y[(size_t)row * GDIM + n0 + colr] = y;
        }
    }
}

// ---------------------------------------------------------------------------
// edge attention score: one wave32 per edge
//   s = leaky_relu(dot(Arows[idxA[e]], attw[0:128]) +
//                  dot(Brows[idxB[e]], attw[128:256]))
// ---------------------------------------------------------------------------
__global__ __launch_bounds__(256)
void edge_score_kernel(const float* __restrict__ Arows, const int* __restrict__ idxA,
                       const float* __restrict__ Brows, const int* __restrict__ idxB,
                       const float* __restrict__ attw, const int* __restrict__ seg,
                       float* __restrict__ sc, unsigned* __restrict__ segmax, int E)
{
    int wid  = (blockIdx.x * blockDim.x + threadIdx.x) >> 5;
    int lane = threadIdx.x & 31;
    if (wid >= E) return;
    int ra = idxA[wid], rb = idxB[wid];
    float4 a  = *(const float4*)(Arows + (size_t)ra * GDIM + lane * 4);
    float4 b  = *(const float4*)(Brows + (size_t)rb * GDIM + lane * 4);
    float4 wa = *(const float4*)(attw + lane * 4);
    float4 wb = *(const float4*)(attw + GDIM + lane * 4);
    float p = a.x * wa.x + a.y * wa.y + a.z * wa.z + a.w * wa.w
            + b.x * wb.x + b.y * wb.y + b.z * wb.z + b.w * wb.w;
#pragma unroll
    for (int off = 16; off > 0; off >>= 1) p += __shfl_down(p, off, 32);
    if (lane == 0) {
        float s = (p > 0.f) ? p : NEG_SLOPE * p;
        sc[wid] = s;
        atomicMax(&segmax[seg[wid]], fmap(s));
    }
}

// ---------------------------------------------------------------------------
// edge exp + denominator: one thread per edge
// ---------------------------------------------------------------------------
__global__ void edge_exp_kernel(const float* __restrict__ sc,
                                const unsigned* __restrict__ segmax,
                                const int* __restrict__ seg,
                                float* __restrict__ exs, float* __restrict__ denom,
                                int E)
{
    int e = blockIdx.x * blockDim.x + threadIdx.x;
    if (e >= E) return;
    int s = seg[e];
    float ex = __expf(sc[e] - funmap(segmax[s]));
    exs[e] = ex;
    atomicAdd(&denom[s], ex);
}

// ---------------------------------------------------------------------------
// weighted scatter-add: one wave32 per edge, lane handles 4 dims
// ---------------------------------------------------------------------------
__global__ __launch_bounds__(256)
void edge_aggr_kernel(const float* __restrict__ msg, const int* __restrict__ srcIdx,
                      const int* __restrict__ dstIdx, const int* __restrict__ seg,
                      const float* __restrict__ exs, const float* __restrict__ denom,
                      float* __restrict__ aggr, int E)
{
    int wid  = (blockIdx.x * blockDim.x + threadIdx.x) >> 5;
    int lane = threadIdx.x & 31;
    if (wid >= E) return;
    float w = exs[wid] / (denom[seg[wid]] + 1e-16f);
    float4 v = *(const float4*)(msg + (size_t)srcIdx[wid] * GDIM + lane * 4);
    float* dst = aggr + (size_t)dstIdx[wid] * GDIM + lane * 4;
    atomicAdd(dst + 0, w * v.x);
    atomicAdd(dst + 1, w * v.y);
    atomicAdd(dst + 2, w * v.z);
    atomicAdd(dst + 3, w * v.w);
}

// ---------------------------------------------------------------------------
// host-side orchestration
// ---------------------------------------------------------------------------
extern "C" void kernel_launch(void* const* d_in, const int* in_sizes, int n_in,
                              void* d_out, int out_size, void* d_ws, size_t ws_size,
                              hipStream_t stream)
{
    const int Lr = in_sizes[4] / GDIM;   // 100000
    const int Cr = in_sizes[5] / GDIM;   // 210000
    const int E  = in_sizes[2];          // 630000
    const int ITERS = 2;

    const int*   l_edge = (const int*)d_in[2];
    const int*   c_edge = (const int*)d_in[3];
    const float* l_emb0 = (const float*)d_in[4];
    const float* c_emb0 = (const float*)d_in[5];
    const float* l2c_w1 = (const float*)d_in[6],  *l2c_b1 = (const float*)d_in[7];
    const float* l2c_w2 = (const float*)d_in[8],  *l2c_b2 = (const float*)d_in[9];
    const float* c2l_w1 = (const float*)d_in[10], *c2l_b1 = (const float*)d_in[11];
    const float* c2l_w2 = (const float*)d_in[12], *c2l_b2 = (const float*)d_in[13];
    const float* l2l_w1 = (const float*)d_in[14], *l2l_b1 = (const float*)d_in[15];
    const float* l2l_w2 = (const float*)d_in[16], *l2l_b2 = (const float*)d_in[17];
    const float* c_att  = (const float*)d_in[18];
    const float* c_upd_w = (const float*)d_in[19], *c_upd_b = (const float*)d_in[20];
    const float* l_att  = (const float*)d_in[21];
    const float* l_upd_w = (const float*)d_in[22], *l_upd_b = (const float*)d_in[23];

    // ---- workspace carve-up (floats), 64B-aligned slices ----
    float* ws = (float*)d_ws;
    size_t off = 0;
    auto carve = [&](size_t n) {
        float* p = ws + off;
        off += (n + 15) & ~(size_t)15;
        return p;
    };
    const size_t MX = (size_t)((Lr > Cr) ? Lr : Cr);
    float* l_msg   = carve((size_t)Lr * GDIM);
    float* c_msg   = carve((size_t)Cr * GDIM);
    float* l2l_msg = carve((size_t)Lr * GDIM);
    float* hidden  = carve(MX * GDIM);
    float* l2c_agg = carve((size_t)Cr * GDIM);
    float* c2l_agg = carve((size_t)Lr * GDIM);
    float* sc      = carve((size_t)E);
    float* exs     = carve((size_t)E);
    unsigned* segmax = (unsigned*)carve((size_t)Cr);
    float* denom   = carve((size_t)Cr);
    _Float16* wh   = (_Float16*)carve(90112);   // 180224 packed-f16 weight halves

    _Float16* w_l2c1 = wh;
    _Float16* w_l2c2 = wh + 1 * 16384;
    _Float16* w_c2l1 = wh + 2 * 16384;
    _Float16* w_c2l2 = wh + 3 * 16384;
    _Float16* w_l2l1 = wh + 4 * 16384;
    _Float16* w_l2l2 = wh + 5 * 16384;
    _Float16* w_cupd = wh + 6 * 16384;            // K=256
    _Float16* w_lupd = wh + 6 * 16384 + 32768;    // K=384

    auto pack = [&](const float* src, _Float16* dst, int K) {
        int n = K * GDIM;
        pack_w_kernel<<<(n + 255) / 256, 256, 0, stream>>>(src, dst, K);
    };
    pack(l2c_w1, w_l2c1, 128); pack(l2c_w2, w_l2c2, 128);
    pack(c2l_w1, w_c2l1, 128); pack(c2l_w2, w_c2l2, 128);
    pack(l2l_w1, w_l2l1, 128); pack(l2l_w2, w_l2l2, 128);
    pack(c_upd_w, w_cupd, 256); pack(l_upd_w, w_lupd, 384);

    // ---- output slices: l_embs[3][L][128] then c_embs[3][C][128] ----
    float* outL = (float*)d_out;
    float* outC = outL + (size_t)3 * Lr * GDIM;
    hipMemcpyAsync(outL, l_emb0, (size_t)Lr * GDIM * sizeof(float),
                   hipMemcpyDeviceToDevice, stream);
    hipMemcpyAsync(outC, c_emb0, (size_t)Cr * GDIM * sizeof(float),
                   hipMemcpyDeviceToDevice, stream);

    auto gemm = [&](const float* x0, const float* x1, const float* x2, int xor0,
                    const _Float16* w, const float* b, float* y,
                    int N, int nsrc, int relu) {
        dim3 g((N + 15) / 16);
        if (nsrc == 1)
            gemm_wmma_kernel<1><<<g, 256, 0, stream>>>(x0, x1, x2, xor0, w, b, y, N, relu);
        else if (nsrc == 2)
            gemm_wmma_kernel<2><<<g, 256, 0, stream>>>(x0, x1, x2, xor0, w, b, y, N, relu);
        else
            gemm_wmma_kernel<3><<<g, 256, 0, stream>>>(x0, x1, x2, xor0, w, b, y, N, relu);
    };
    auto zero = [&](void* p, long long nwords) {
        zero32_kernel<<<(unsigned)((nwords + 255) / 256), 256, 0, stream>>>(
            (unsigned*)p, nwords);
    };
    const int egrid = (E + 7) / 8;   // wave-per-edge kernels, 8 waves / block

    const float* lcur = outL;
    const float* ccur = outC;
    for (int t = 0; t < ITERS; ++t) {
        float* lnext = outL + (size_t)(t + 1) * Lr * GDIM;
        float* cnext = outC + (size_t)(t + 1) * Cr * GDIM;

        // message MLPs (2-layer, relu between)
        gemm(lcur, nullptr, nullptr, 0, w_l2c1, l2c_b1, hidden, Lr, 1, 1);
        gemm(hidden, nullptr, nullptr, 0, w_l2c2, l2c_b2, l_msg, Lr, 1, 0);
        gemm(ccur, nullptr, nullptr, 0, w_c2l1, c2l_b1, hidden, Cr, 1, 1);
        gemm(hidden, nullptr, nullptr, 0, w_c2l2, c2l_b2, c_msg, Cr, 1, 0);
        // l2l: input row j is l_emb[j^1]
        gemm(lcur, nullptr, nullptr, 1, w_l2l1, l2l_b1, hidden, Lr, 1, 1);
        gemm(hidden, nullptr, nullptr, 0, w_l2l2, l2l_b2, l2l_msg, Lr, 1, 0);

        // ---- literal -> clause attention (softmax over c_edge segments) ----
        zero(segmax, 2LL * Cr);        // segmax + denom (adjacent carves)
        zero(l2c_agg, (long long)Cr * GDIM);
        edge_score_kernel<<<egrid, 256, 0, stream>>>(
            ccur, c_edge, l_msg, l_edge, c_att, c_edge, sc, segmax, E);
        edge_exp_kernel<<<(E + 255) / 256, 256, 0, stream>>>(
            sc, segmax, c_edge, exs, denom, E);
        edge_aggr_kernel<<<egrid, 256, 0, stream>>>(
            l_msg, l_edge, c_edge, c_edge, exs, denom, l2c_agg, E);

        // ---- clause -> literal attention (softmax over c_edge: faithful) ----
        zero(segmax, 2LL * Cr);
        zero(c2l_agg, (long long)Lr * GDIM);
        edge_score_kernel<<<egrid, 256, 0, stream>>>(
            lcur, l_edge, c_msg, c_edge, l_att, c_edge, sc, segmax, E);
        edge_exp_kernel<<<(E + 255) / 256, 256, 0, stream>>>(
            sc, segmax, c_edge, exs, denom, E);
        edge_aggr_kernel<<<egrid, 256, 0, stream>>>(
            c_msg, c_edge, l_edge, c_edge, exs, denom, c2l_agg, E);

        // ---- updates (concat GEMMs) ----
        gemm(ccur, l2c_agg, nullptr, 0, w_cupd, c_upd_b, cnext, Cr, 2, 0);
        gemm(lcur, c2l_agg, l2l_msg, 0, w_lupd, l_upd_b, lnext, Lr, 3, 0);

        lcur = lnext;
        ccur = cnext;
    }
}
